// TopologicalDiversitySampler_11845519802580
// MI455X (gfx1250) — matrix operand order repaired and checked by reading, not compiled, not run on hardware
//
#include <hip/hip_runtime.h>
#include <hip/hip_fp16.h>
#include <math.h>

typedef __attribute__((ext_vector_type(8)))  _Float16 v8h;
typedef __attribute__((ext_vector_type(16))) _Float16 v16h;
typedef __attribute__((ext_vector_type(8)))  float    v8f;

__device__ __forceinline__ float readlane_f(float v, int lane) {
  return __builtin_bit_cast(float,
           __builtin_amdgcn_readlane(__builtin_bit_cast(int, v), lane));
}

// Swizzled f16 feature store, WMMA A-fragment order (D fixed at 256):
//   halves offset(tile, chunk j, slot) = tile*4096 + j*512 + slot*16
//   slot s<16 : row tile*16+s, halves [K j*32+0..7 | K j*32+16..23]
//   slot s>=16: row tile*16+s-16, halves [K j*32+8..15 | K j*32+24..31]
// => lane l loads ONE contiguous v16h per chunk; a wave reads 1KB contiguous.

// ---------------------------------------------------------------------------
// K0: reduce attention -> a_min, a_max, argmax (first occurrence). 1 block.
// hdr words: [0]=a_min(f32) [1]=a_max(f32) [2]=first(i32) [3]=cur(i32)
// ---------------------------------------------------------------------------
__global__ __launch_bounds__(1024)
void k_attn_reduce(const float* __restrict__ attn, int N,
                   float* __restrict__ hdr_f, int* __restrict__ hdr_i,
                   float* __restrict__ out) {
  __shared__ float smn[1024], smx[1024], sbv[1024];
  __shared__ int   sbi[1024];
  int t = threadIdx.x;
  float mn = INFINITY, mx = -INFINITY, bv = -INFINITY;
  int bi = 0x7fffffff;
  for (int i = t; i < N; i += 1024) {
    float v = attn[i];
    mn = fminf(mn, v);
    mx = fmaxf(mx, v);
    if (v > bv || (v == bv && i < bi)) { bv = v; bi = i; }
  }
  smn[t] = mn; smx[t] = mx; sbv[t] = bv; sbi[t] = bi;
  __syncthreads();
  for (int off = 512; off > 0; off >>= 1) {
    if (t < off) {
      smn[t] = fminf(smn[t], smn[t + off]);
      smx[t] = fmaxf(smx[t], smx[t + off]);
      float ov = sbv[t + off]; int oi = sbi[t + off];
      if (ov > sbv[t] || (ov == sbv[t] && oi < sbi[t])) { sbv[t] = ov; sbi[t] = oi; }
    }
    __syncthreads();
  }
  if (t == 0) {
    hdr_f[0] = smn[0];
    hdr_f[1] = smx[0];
    hdr_i[2] = sbi[0];
    hdr_i[3] = sbi[0];           // cur_idx for iteration 1
    out[0]   = (float)sbi[0];    // first selected index
  }
}

// ---------------------------------------------------------------------------
// K1: attn_norm = minmax-normalized attention; -inf for pad rows and for the
// first selected point (doubles as the "selected" mask). min_d = +inf.
// ---------------------------------------------------------------------------
__global__ __launch_bounds__(256)
void k_init(const float* __restrict__ attn, int N, int Npad,
            const float* __restrict__ hdr_f, const int* __restrict__ hdr_i,
            float* __restrict__ attn_n, float* __restrict__ min_d) {
  int i = blockIdx.x * blockDim.x + threadIdx.x;
  if (i >= Npad) return;
  float a_min = hdr_f[0], a_max = hdr_f[1];
  int first = hdr_i[2];
  float v = -INFINITY;
  if (i < N && i != first) v = (attn[i] - a_min) / (a_max - a_min + 1e-10f);
  attn_n[i] = v;
  min_d[i]  = INFINITY;
}

// ---------------------------------------------------------------------------
// K2: L2-normalize each row, convert to f16, and store in swizzled WMMA
// A-fragment order. One wave per row; the row's 256 elements split into 32
// groups of 8 contiguous K values -> one group per lane: contiguous 32B source
// read + one 16B store. (f16 halves traffic; 51MB stays L2-resident.)
// ---------------------------------------------------------------------------
__global__ __launch_bounds__(256)
void k_normalize_sw(const float* __restrict__ feats, int N, int Npad,
                    _Float16* __restrict__ sw) {
  const int wid = threadIdx.x >> 5, lane = threadIdx.x & 31;
  const int row = blockIdx.x * 8 + wid;
  if (row >= Npad) return;
  const int tile = row >> 4, s = row & 15;
  const int j = lane >> 2, g = lane & 3;
  // g=0 -> K j*32+0..7   -> slot s,    idx 0..7
  // g=1 -> K j*32+8..15  -> slot s+16, idx 0..7
  // g=2 -> K j*32+16..23 -> slot s,    idx 8..15
  // g=3 -> K j*32+24..31 -> slot s+16, idx 8..15
  const int kbase = j * 32 + ((g & 1) ? 8 : 0) + ((g & 2) ? 16 : 0);
  const size_t dstOff = (size_t)tile * 4096 + (size_t)j * 512 +
                        (size_t)(s + ((g & 1) ? 16 : 0)) * 16 +
                        ((g & 2) ? 8 : 0);
  if (row < N) {
    const float* src = feats + (size_t)row * 256;
    float ss = 0.f;
    for (int e = lane; e < 256; e += 32) { float f = src[e]; ss += f * f; }
    #pragma unroll
    for (int off = 16; off; off >>= 1) ss += __shfl_xor(ss, off, 32);
    float inv = 1.0f / fmaxf(sqrtf(ss), 1e-12f);
    v8h h;
    #pragma unroll
    for (int i = 0; i < 8; ++i) h[i] = (_Float16)(src[kbase + i] * inv);
    *(v8h*)(sw + dstOff) = h;
  } else {
    v8h z = {};
    *(v8h*)(sw + dstOff) = z;
  }
}

// ---------------------------------------------------------------------------
// K3: one FPS step. Each wave owns 16-point tiles; dots vs. the query row via
// 8x v_wmma_f32_16x16x32_f16 (query replicated in all 16 B columns -> every C
// column equals the per-point dot). sched_barrier(0) pins all 16 b128 loads
// (8KB/wave in flight) ahead of the WMMA burst; next tile is prefetched while
// this one computes. Dot extraction via v_readlane broadcasts from lanes 0/16.
// ---------------------------------------------------------------------------
__global__ __launch_bounds__(256)
void k_step(const _Float16* __restrict__ sw, const float* __restrict__ attn_n,
            float* __restrict__ min_d, const int* __restrict__ hdr_i,
            float* __restrict__ bres_v, int* __restrict__ bres_i,
            int numTiles) {
  const int lane = threadIdx.x & 31;
  const int wid  = threadIdx.x >> 5;
  const int hi   = (lane >> 4) & 1;
  const int col  = lane & 15;
  const int nW   = blockDim.x >> 5;
  const int gw   = blockIdx.x * nW + wid;
  const int gstr = gridDim.x * nW;

  // Reconstruct the query row (B fragment) from swizzled storage.
  // lanes<16 need [K j*32+0..7 | +16..23]; lanes>=16 need [K j*32+8..15 | +24..31].
  const int cur = hdr_i[3];
  const int qt = cur >> 4, qs = cur & 15;
  v16h b[8];
  #pragma unroll
  for (int j = 0; j < 8; ++j) {
    const _Float16* b0 = sw + (size_t)qt * 4096 + j * 512 + qs * 16;
    v8h lo = *(const v8h*)(b0 + hi * 256);        // K +0..7  or +8..15
    v8h hh = *(const v8h*)(b0 + hi * 256 + 8);    // K +16..23 or +24..31
    b[j] = __builtin_shufflevector(lo, hh,
             0,1,2,3,4,5,6,7,8,9,10,11,12,13,14,15);
  }

  float bestV = -INFINITY;
  int   bestI = 0x7fffffff;

  // Per-lane A pointer: contiguous 32B per chunk, 8KB contiguous per tile.
  const _Float16* arp = sw + (size_t)gw * 4096 + lane * 16;
  const size_t arStride = (size_t)gstr * 4096;
  // Per-lane prefetch pointer covering the NEXT tile (lane*256B + 2 lines).
  const char* pfp = (const char*)sw + ((size_t)(gw + gstr)) * 8192 + lane * 256;
  const size_t pfStride = (size_t)gstr * 8192;

  for (int tile = gw; tile < numTiles; tile += gstr,
       arp += arStride, pfp += pfStride) {
    const int base = tile * 16;

    v16h a[8];
    #pragma unroll
    for (int j = 0; j < 8; ++j)
      a[j] = *(const v16h*)(arp + j * 512);

    if (tile + gstr < numTiles) {        // wave-uniform
      __builtin_prefetch(pfp, 0, 3);
      __builtin_prefetch(pfp + 128, 0, 3);
    }

    __builtin_amdgcn_sched_barrier(0);   // loads stay above, WMMAs below

    v8f c = {};
    #pragma unroll
    for (int j = 0; j < 8; ++j)
      c = __builtin_amdgcn_wmma_f32_16x16x32_f16(
              false, a[j], false, b[j], (short)0, c, false, false);

    // C layout: VGPR r = row r (lanes 0-15) / row 8+r (lanes 16-31); all
    // columns identical. Broadcast via v_readlane; lane L(<16) owns base+L.
    float dot = 0.f;
    #pragma unroll
    for (int r = 0; r < 8; ++r) {
      float v0 = readlane_f(c[r], 0);    // dot of point base+r
      float v1 = readlane_f(c[r], 16);   // dot of point base+8+r
      dot = (col == r)     ? v0 : dot;
      dot = (col == 8 + r) ? v1 : dot;
    }
    if (lane < 16) {
      int p = base + lane;
      float dist = sqrtf(fmaxf(2.0f - 2.0f * dot, 0.0f));
      float m = fminf(min_d[p], dist);
      min_d[p] = m;
      float comb = 0.5f * attn_n[p] + 0.5f * m;   // -inf if selected/pad
      if (comb > bestV || (comb == bestV && p < bestI)) { bestV = comb; bestI = p; }
    }
  }

  // Wave32 butterfly argmax (ties -> smaller index).
  #pragma unroll
  for (int off = 16; off; off >>= 1) {
    float ov = __shfl_xor(bestV, off, 32);
    int   oi = __shfl_xor(bestI, off, 32);
    if (ov > bestV || (ov == bestV && oi < bestI)) { bestV = ov; bestI = oi; }
  }
  __shared__ float sv[8];
  __shared__ int   si[8];
  if (lane == 0) { sv[wid] = bestV; si[wid] = bestI; }
  __syncthreads();
  if (threadIdx.x == 0) {
    float bv = sv[0]; int bi = si[0];
    for (int w = 1; w < nW; ++w)
      if (sv[w] > bv || (sv[w] == bv && si[w] < bi)) { bv = sv[w]; bi = si[w]; }
    bres_v[blockIdx.x] = bv;
    bres_i[blockIdx.x] = bi;
  }
}

// ---------------------------------------------------------------------------
// K4: final argmax over per-block partials; mark selected (attn_n = -inf),
// store cur_idx for the next step's query, emit the index.
// ---------------------------------------------------------------------------
__global__ __launch_bounds__(256)
void k_pick(const float* __restrict__ bres_v, const int* __restrict__ bres_i,
            int nBlocks, float* __restrict__ attn_n, int* __restrict__ hdr_i,
            float* __restrict__ out, int t) {
  __shared__ float sv[256];
  __shared__ int   si[256];
  int tid = threadIdx.x;
  float bv = -INFINITY; int bi = 0x7fffffff;
  for (int i = tid; i < nBlocks; i += blockDim.x) {
    float v = bres_v[i]; int ix = bres_i[i];
    if (v > bv || (v == bv && ix < bi)) { bv = v; bi = ix; }
  }
  sv[tid] = bv; si[tid] = bi;
  __syncthreads();
  for (int off = 128; off; off >>= 1) {
    if (tid < off) {
      if (sv[tid + off] > sv[tid] ||
          (sv[tid + off] == sv[tid] && si[tid + off] < si[tid])) {
        sv[tid] = sv[tid + off]; si[tid] = si[tid + off];
      }
    }
    __syncthreads();
  }
  if (tid == 0) {
    int best = si[0];
    attn_n[best] = -INFINITY;   // selected mask
    hdr_i[3]     = best;        // query for next iteration
    out[t]       = (float)best;
  }
}

// ---------------------------------------------------------------------------
extern "C" void kernel_launch(void* const* d_in, const int* in_sizes, int n_in,
                              void* d_out, int out_size, void* d_ws, size_t ws_size,
                              hipStream_t stream) {
  const float* feats = (const float*)d_in[0];
  const float* attn  = (const float*)d_in[1];
  const int N = in_sizes[1];
  const int k = out_size;             // k (device scalar mirrored by out_size)
  const int Npad = ((N + 127) / 128) * 128;

  // Workspace layout (all 64B aligned):
  //   [0,64)            header: a_min, a_max, first_idx, cur_idx
  //   [64, 64+2048)     per-block best values (float)
  //   [64+2048, +4096)  per-block best indices (int)
  //   attn_n  float[Npad]
  //   min_d   float[Npad]
  //   sw      f16[Npad*256]  swizzled A-fragments (~51 MB -> L2-resident)
  char* ws = (char*)d_ws;
  float*    hdr_f  = (float*)ws;
  int*      hdr_i  = (int*)ws;
  float*    bres_v = (float*)(ws + 64);
  int*      bres_i = (int*)(ws + 64 + 2048);
  float*    attn_n = (float*)(ws + 64 + 4096);
  float*    min_d  = (float*)((char*)attn_n + (size_t)Npad * 4);
  _Float16* sw     = (_Float16*)((char*)min_d + (size_t)Npad * 4);
  float*    out    = (float*)d_out;

  k_attn_reduce<<<1, 1024, 0, stream>>>(attn, N, hdr_f, hdr_i, out);
  k_init<<<(Npad + 255) / 256, 256, 0, stream>>>(attn, N, Npad, hdr_f, hdr_i,
                                                 attn_n, min_d);
  k_normalize_sw<<<Npad / 8, 256, 0, stream>>>(feats, N, Npad, sw);

  const int numTiles = Npad / 16;
  const int STEP_BLOCKS = 256;   // 2048 waves, ~3 tiles each
  for (int t = 1; t < k; ++t) {
    k_step<<<STEP_BLOCKS, 256, 0, stream>>>(sw, attn_n, min_d, hdr_i,
                                            bres_v, bres_i, numTiles);
    k_pick<<<1, 256, 0, stream>>>(bres_v, bres_i, STEP_BLOCKS, attn_n, hdr_i,
                                  out, t);
  }
}
